// DropConnectLSTMCell_4879082849038
// MI455X (gfx1250) — matrix operand order repaired and checked by reading, not compiled, not run on hardware
//
#include <hip/hip_runtime.h>

typedef __attribute__((ext_vector_type(16))) _Float16 v16h;
typedef __attribute__((ext_vector_type(8)))  _Float16 v8h;
typedef __attribute__((ext_vector_type(4)))  _Float16 v4h;
typedef __attribute__((ext_vector_type(8)))  float    v8f;

// Problem sizes (fixed by the reference)
constexpr int ROWS  = 8192;   // batch
constexpr int DIN   = 1024;   // input features (== units, so X and H strides match)
constexpr int NU    = 1024;   // units
constexpr int NCOL  = 4096;   // 4 * units (gate columns of the weight matrices)

constexpr float DROP_RATE  = 0.1f;             // dropout & recurrent dropout
constexpr float KDROP_RATE = 0.05f;            // kernel & recurrent-kernel DropConnect
constexpr float A_SCALE    = 1.0f / (1.0f - DROP_RATE);
constexpr float W_SCALE    = 1.0f / (1.0f - KDROP_RATE);

// Tiling
constexpr int BM  = 128;      // rows per block
constexpr int BU  = 32;       // units per block (=> 4*BU = 128 gate columns)
constexpr int BK  = 32;       // K per step (one WMMA K depth)
constexpr int LDA = BK + 8;   // padded LDS stride (halves) for A tile
constexpr int LDB = BK + 8;   // padded LDS stride (halves) for B tile
constexpr int ASZ = BM * LDA;        // one A buffer (halves)
constexpr int BSZ = 4 * BU * LDB;    // one B buffer (halves)
constexpr int STEPS_PER_PHASE = DIN / BK;   // 32

__device__ __forceinline__ float sigmoidf_(float z) {
    return 1.0f / (1.0f + __expf(-z));
}

__device__ __forceinline__ v16h join16(v8h lo, v8h hi) {
    return __builtin_shufflevector(lo, hi, 0, 1, 2, 3, 4, 5, 6, 7,
                                           8, 9, 10, 11, 12, 13, 14, 15);
}

__global__ __launch_bounds__(256)
void dclstm_wmma_f16(const float* __restrict__ X,     // [ROWS, DIN]
                     const float* __restrict__ H,     // [ROWS, NU]
                     const float* __restrict__ C,     // [ROWS, NU]
                     const float* __restrict__ W,     // [DIN, NCOL]
                     const float* __restrict__ RW,    // [NU, NCOL]
                     const float* __restrict__ BIAS,  // [NCOL]
                     const float* __restrict__ DPU,   // [ROWS, DIN]
                     const float* __restrict__ RDPU,  // [ROWS, NU]
                     const float* __restrict__ KDPU,  // [DIN, NCOL]
                     const float* __restrict__ RKDPU, // [NU, NCOL]
                     float* __restrict__ Hn,          // [ROWS, NU]
                     float* __restrict__ Cn)          // [ROWS, NU]
{
    __shared__ _Float16 As[2 * ASZ];     // A tiles: 128 rows x 32 K (row-major, K contiguous)
    __shared__ _Float16 Bs[2 * BSZ];     // B tiles: 128 cols x 32 K (col-major, K contiguous)

    const int tid  = threadIdx.x;
    const int lane = tid & 31;
    const int wave = tid >> 5;
    const int wm   = wave & 3;               // wave's M sub-tile (0..3)  -> 32 rows each
    const int wu   = wave >> 2;              // wave's unit sub-tile (0..1) -> 16 units each

    const int m0 = blockIdx.x * BM;          // row base of this block
    const int u0 = blockIdx.y * BU;          // unit base of this block

    // Cooperative-load coordinates
    const int arow = tid >> 3;               // 0..31  (A: 32 rows / pass)
    const int ak4  = (tid & 7) * 4;          // 0..28  (A: 4 K-elems per thread)
    const int bkk  = tid >> 5;               // 0..7   (B: 8 K-rows / pass)
    const int bc4  = (tid & 31) * 4;         // 0..124 (B: 4 cols per thread)
    const int bg   = bc4 >> 5;               // gate of this thread's B columns
    const int buu  = bc4 & 31;               // unit-in-block of this thread's B columns

    // Per-thread global base offsets (element units); X/H and W/RW share strides.
    const long aBase = (long)(m0 + arow) * DIN + ak4;
    const long bBase = (long)bkk * NCOL + (long)bg * NU + u0 + buu;

    // WMMA fragment constants (ISA 16-bit A/B VGPR layouts)
    const int ka = (lane >> 4) * 8;          // A: K-chunk base (0 or 8), second chunk at +16
    const int kc = (lane >> 4) * 8;          // B: K-chunk base (0 or 8), second chunk at +16
    const int ln = lane & 15;
    const int aOff0 = (wm * 32 + ln) * LDA;
    const int aOff1 = (wm * 32 + 16 + ln) * LDA;

    v8f acc[4][2];
    #pragma unroll
    for (int g = 0; g < 4; ++g)
        #pragma unroll
        for (int t = 0; t < 2; ++t)
            acc[g][t] = (v8f)(0.0f);

    // ---- stage one 128x32 A tile + 32x128 B tile into an LDS buffer ----
    auto stage = [&](const float* av, const float* am,
                     const float* bv, const float* bmk,
                     _Float16* Asb, _Float16* Bsb, bool pf) {
        #pragma unroll
        for (int p = 0; p < 4; ++p) {
            const float4 xv = *(const float4*)(av + (long)p * 32 * DIN);
            const float4 uv = *(const float4*)(am + (long)p * 32 * DIN);
            v4h o;
            o[0] = (_Float16)(xv.x * (uv.x >= DROP_RATE ? A_SCALE : 0.0f));
            o[1] = (_Float16)(xv.y * (uv.y >= DROP_RATE ? A_SCALE : 0.0f));
            o[2] = (_Float16)(xv.z * (uv.z >= DROP_RATE ? A_SCALE : 0.0f));
            o[3] = (_Float16)(xv.w * (uv.w >= DROP_RATE ? A_SCALE : 0.0f));
            *(v4h*)(&Asb[(arow + p * 32) * LDA + ak4]) = o;
        }
        #pragma unroll
        for (int p = 0; p < 4; ++p) {
            const float4 wv = *(const float4*)(bv + (long)p * 8 * NCOL);
            const float4 uv = *(const float4*)(bmk + (long)p * 8 * NCOL);
            const int kk = bkk + p * 8;
            Bsb[(bc4 + 0) * LDB + kk] = (_Float16)(wv.x * (uv.x >= KDROP_RATE ? W_SCALE : 0.0f));
            Bsb[(bc4 + 1) * LDB + kk] = (_Float16)(wv.y * (uv.y >= KDROP_RATE ? W_SCALE : 0.0f));
            Bsb[(bc4 + 2) * LDB + kk] = (_Float16)(wv.z * (uv.z >= KDROP_RATE ? W_SCALE : 0.0f));
            Bsb[(bc4 + 3) * LDB + kk] = (_Float16)(wv.w * (uv.w >= KDROP_RATE ? W_SCALE : 0.0f));
        }
        if (pf) {  // pull tile i+2 toward L2 (global_prefetch_b8; speculative => safe)
            __builtin_prefetch(av + BK, 0, 3);
            __builtin_prefetch(bv + (long)BK * NCOL, 0, 3);
        }
    };

    // ---- 8 WMMAs on the given LDS buffer ----
    auto compute = [&](const _Float16* Asb, const _Float16* Bsb) {
        v16h aF[2];
        aF[0] = join16(*(const v8h*)(Asb + aOff0 + ka),
                       *(const v8h*)(Asb + aOff0 + ka + 16));
        aF[1] = join16(*(const v8h*)(Asb + aOff1 + ka),
                       *(const v8h*)(Asb + aOff1 + ka + 16));
        #pragma unroll
        for (int g = 0; g < 4; ++g) {
            const _Float16* bp = Bsb + (g * 32 + wu * 16 + ln) * LDB + kc;
            const v16h bF = join16(*(const v8h*)bp, *(const v8h*)(bp + 16));
            acc[g][0] = __builtin_amdgcn_wmma_f32_16x16x32_f16(
                false, aF[0], false, bF, (short)0, acc[g][0], false, false);
            acc[g][1] = __builtin_amdgcn_wmma_f32_16x16x32_f16(
                false, aF[1], false, bF, (short)0, acc[g][1], false, false);
        }
    };

    // ================= pipelined main loop (double-buffered LDS) =================
    int buf = 0;

    // Prologue: stage X tile 0 into buffer 0.
    stage(X + aBase, DPU + aBase, W + bBase, KDPU + bBase, As, Bs, true);

    // Phase 1: compute X tiles 0..30, staging X tiles 1..31.
    {
        const float* aV = X    + aBase + BK;
        const float* aM = DPU  + aBase + BK;
        const float* bV = W    + bBase + (long)BK * NCOL;
        const float* bM = KDPU + bBase + (long)BK * NCOL;
        for (int i = 0; i < STEPS_PER_PHASE - 1; ++i) {
            __syncthreads();
            stage(aV, aM, bV, bM, As + (buf ^ 1) * ASZ, Bs + (buf ^ 1) * BSZ,
                  i < STEPS_PER_PHASE - 2);
            aV += BK; aM += BK;
            bV += (long)BK * NCOL; bM += (long)BK * NCOL;
            compute(As + buf * ASZ, Bs + buf * BSZ);
            buf ^= 1;
        }
    }

    // Boundary: stage H tile 0, compute last X tile.
    __syncthreads();
    stage(H + aBase, RDPU + aBase, RW + bBase, RKDPU + bBase,
          As + (buf ^ 1) * ASZ, Bs + (buf ^ 1) * BSZ, true);
    compute(As + buf * ASZ, Bs + buf * BSZ);
    buf ^= 1;

    // Phase 2: compute H tiles 0..30, staging H tiles 1..31.
    {
        const float* aV = H     + aBase + BK;
        const float* aM = RDPU  + aBase + BK;
        const float* bV = RW    + bBase + (long)BK * NCOL;
        const float* bM = RKDPU + bBase + (long)BK * NCOL;
        for (int i = 0; i < STEPS_PER_PHASE - 1; ++i) {
            __syncthreads();
            stage(aV, aM, bV, bM, As + (buf ^ 1) * ASZ, Bs + (buf ^ 1) * BSZ,
                  i < STEPS_PER_PHASE - 2);
            aV += BK; aM += BK;
            bV += (long)BK * NCOL; bM += (long)BK * NCOL;
            compute(As + buf * ASZ, Bs + buf * BSZ);
            buf ^= 1;
        }
    }

    // Epilogue of the pipeline: compute last H tile.
    __syncthreads();
    compute(As + buf * ASZ, Bs + buf * BSZ);

    // ---- fused gate epilogue: all 4 gates of (row, unit) live in this lane ----
    const int unit = u0 + wu * 16 + ln;
    const float bI = BIAS[0 * NU + unit];
    const float bF = BIAS[1 * NU + unit];
    const float bC = BIAS[2 * NU + unit];
    const float bO = BIAS[3 * NU + unit];
    const int mHi = (lane >> 4) * 8;         // C layout: m = (lane/16)*8 + vgpr

    #pragma unroll
    for (int t = 0; t < 2; ++t) {
        #pragma unroll
        for (int r = 0; r < 8; ++r) {
            const int  row = m0 + wm * 32 + t * 16 + mHi + r;
            const long o   = (long)row * NU + unit;
            const float zi = acc[0][t][r] + bI;
            const float zf = acc[1][t][r] + bF;
            const float zc = acc[2][t][r] + bC;
            const float zo = acc[3][t][r] + bO;
            const float ig = sigmoidf_(zi);
            const float fg = sigmoidf_(zf);
            const float og = sigmoidf_(zo);
            const float cn = fg * C[o] + ig * tanhf(zc);
            Hn[o] = og * tanhf(cn);
            Cn[o] = cn;
        }
    }
}

extern "C" void kernel_launch(void* const* d_in, const int* in_sizes, int n_in,
                              void* d_out, int out_size, void* d_ws, size_t ws_size,
                              hipStream_t stream) {
    (void)in_sizes; (void)n_in; (void)out_size; (void)d_ws; (void)ws_size;
    const float* X     = (const float*)d_in[0];
    const float* H     = (const float*)d_in[1];
    const float* C     = (const float*)d_in[2];
    const float* W     = (const float*)d_in[3];
    const float* RW    = (const float*)d_in[4];
    const float* BIAS  = (const float*)d_in[5];
    const float* DPU   = (const float*)d_in[6];
    const float* RDPU  = (const float*)d_in[7];
    const float* KDPU  = (const float*)d_in[8];
    const float* RKDPU = (const float*)d_in[9];

    float* Hn = (float*)d_out;
    float* Cn = (float*)d_out + (long)ROWS * NU;

    dim3 grid(ROWS / BM, NU / BU);   // 64 x 32 blocks
    dim3 block(256);
    dclstm_wmma_f16<<<grid, block, 0, stream>>>(X, H, C, W, RW, BIAS,
                                                DPU, RDPU, KDPU, RKDPU, Hn, Cn);
}
